// WaveRNN_85950885527647
// MI455X (gfx1250) — matrix operand order repaired and checked by reading, not compiled, not run on hardware
//
#include <hip/hip_runtime.h>

#define H  896
#define HH 448
#define NC 256
#define BB 16384

typedef __attribute__((ext_vector_type(16))) __bf16 v16bf;
typedef __attribute__((ext_vector_type(8)))  __bf16 v8bf;
typedef __attribute__((ext_vector_type(8)))  float  v8f;

union Frag { v16bf v; v8bf h[2]; };

__device__ __forceinline__ __bf16 f2bf(float f) {
  union { float f; unsigned u; } c; c.f = f;
  unsigned r = (c.u + 0x7FFFu + ((c.u >> 16) & 1u)) >> 16;
  union { unsigned short s; __bf16 b; } o; o.s = (unsigned short)r;
  return o.b;
}

// CDNA5 async global -> LDS copy (16B per lane), tracked by ASYNCcnt.
__device__ __forceinline__ void async_copy_b128(unsigned lds_off, const void* g) {
  asm volatile("global_load_async_to_lds_b128 %0, %1, off"
               :: "v"(lds_off), "v"((unsigned long long)(uintptr_t)g)
               : "memory");
}
__device__ __forceinline__ void wait_async0() {
  asm volatile("s_wait_asynccnt 0x0" ::: "memory");
}

__device__ __forceinline__ v8f wmma_bf16(const Frag& a, const Frag& b, v8f c) {
  return __builtin_amdgcn_wmma_f32_16x16x32_bf16(false, a.v, false, b.v,
                                                 (short)0, c, false, false);
}

// ---------------- fp32 -> bf16 conversion ----------------
__global__ void cvt_f32_bf16(const float* __restrict__ src,
                             __bf16* __restrict__ dst, int n) {
  int i = blockIdx.x * blockDim.x + threadIdx.x;
  if (i < n) dst[i] = f2bf(src[i]);
}

// ---------------- Stage 1: fused GRU cell ----------------
// grid: (BB/32, H/64), block: 128 (4 waves).
// Block stages a 32-row x 896 bf16 A tile in LDS via async copy; wave w owns
// column tile j=(blockIdx.y*4+w)*16 and computes u/r/e x {2 row tiles}
// (6 accumulators) sharing each B fragment across both row tiles.
#define S1_STRIDE 904   // 896 + 8 bf16 pad: spreads LDS banks
__global__ __launch_bounds__(128)
void gru_stage1(const __bf16* __restrict__ ht1b,   // BB x H  (bf16)
                const __bf16* __restrict__ Rwb,    // 3H x H  (bf16)
                const float*  __restrict__ ht1f,   // BB x H  (fp32)
                const float*  __restrict__ ct_1,
                const float*  __restrict__ ft_1,
                const float*  __restrict__ ct,
                const float*  __restrict__ Icw,    // 3HH x 2
                const float*  __restrict__ Ifw,    // 3HH x 3
                const float*  __restrict__ bu,
                const float*  __restrict__ br,
                const float*  __restrict__ be,
                float*  __restrict__ ht_out,       // BB x H (fp32, d_out)
                __bf16* __restrict__ htb)          // BB x H (bf16, ws)
{
  __shared__ __bf16 atile[32 * S1_STRIDE];         // 57,856 B
  const int lane  = threadIdx.x & 31;
  const int wave  = threadIdx.x >> 5;
  const int hhalf = lane >> 4;
  const int l16   = lane & 15;
  const int rbase = blockIdx.x * 32;
  const int j     = (blockIdx.y * 4 + wave) * 16;

  // Cooperative async stage of A tile: 32 rows x 112 16B-chunks = 3584 chunks,
  // 28 per thread.
  {
    const int t = threadIdx.x;
#pragma unroll 4
    for (int i = 0; i < 28; ++i) {
      const int c    = t + i * 128;
      const int row  = c / 112;
      const int col8 = (c % 112) * 8;            // element offset
      const unsigned loff =
          (unsigned)(uintptr_t)&atile[row * S1_STRIDE + col8];
      async_copy_b128(loff, ht1b + (size_t)(rbase + row) * H + col8);
    }
  }
  wait_async0();
  __syncthreads();

  v8f aU0 = {}, aR0 = {}, aE0 = {}, aU1 = {}, aR1 = {}, aE1 = {};
  const __bf16* a0 = atile + (size_t)l16 * S1_STRIDE;
  const __bf16* a1 = atile + (size_t)(16 + l16) * S1_STRIDE;
  const __bf16* bU = Rwb + (size_t)(j + l16) * H;
  const __bf16* bR = Rwb + (size_t)(H + j + l16) * H;
  const __bf16* bE = Rwb + (size_t)(2 * H + j + l16) * H;

  for (int k = 0; k < H; k += 32) {
    Frag A0, A1, fu, fr, fe;
    A0.h[0] = *(const v8bf*)(a0 + k + 8 * hhalf);
    A0.h[1] = *(const v8bf*)(a0 + k + 16 + 8 * hhalf);
    A1.h[0] = *(const v8bf*)(a1 + k + 8 * hhalf);
    A1.h[1] = *(const v8bf*)(a1 + k + 16 + 8 * hhalf);
    const int bo = k + 8 * hhalf;
    fu.h[0] = *(const v8bf*)(bU + bo);  fu.h[1] = *(const v8bf*)(bU + bo + 16);
    fr.h[0] = *(const v8bf*)(bR + bo);  fr.h[1] = *(const v8bf*)(bR + bo + 16);
    fe.h[0] = *(const v8bf*)(bE + bo);  fe.h[1] = *(const v8bf*)(bE + bo + 16);
    aU0 = wmma_bf16(A0, fu, aU0);  aU1 = wmma_bf16(A1, fu, aU1);
    aR0 = wmma_bf16(A0, fr, aR0);  aR1 = wmma_bf16(A1, fr, aR1);
    aE0 = wmma_bf16(A0, fe, aE0);  aE1 = wmma_bf16(A1, fe, aE1);
  }

  // Rank-2/3 input-projection weights for output column jj (region split is
  // wave-uniform: a 16-col tile never straddles HH).
  const int jj = j + l16;
  float wU0, wU1, wU2, wR0, wR1, wR2, wE0, wE1, wE2;
  if (jj < HH) {
    wU0 = Icw[jj * 2 + 0];            wU1 = Icw[jj * 2 + 1];            wU2 = 0.f;
    wR0 = Icw[(HH + jj) * 2 + 0];     wR1 = Icw[(HH + jj) * 2 + 1];     wR2 = 0.f;
    wE0 = Icw[(2 * HH + jj) * 2 + 0]; wE1 = Icw[(2 * HH + jj) * 2 + 1]; wE2 = 0.f;
  } else {
    const int q = jj - HH;
    wU0 = Ifw[q * 3 + 0];            wU1 = Ifw[q * 3 + 1];            wU2 = Ifw[q * 3 + 2];
    wR0 = Ifw[(HH + q) * 3 + 0];     wR1 = Ifw[(HH + q) * 3 + 1];     wR2 = Ifw[(HH + q) * 3 + 2];
    wE0 = Ifw[(2 * HH + q) * 3 + 0]; wE1 = Ifw[(2 * HH + q) * 3 + 1]; wE2 = Ifw[(2 * HH + q) * 3 + 2];
  }
  const float vbu = bu[jj], vbr = br[jj], vbe = be[jj];

#pragma unroll
  for (int tile = 0; tile < 2; ++tile) {
    const v8f& aU = tile ? aU1 : aU0;
    const v8f& aR = tile ? aR1 : aR0;
    const v8f& aE = tile ? aE1 : aE0;
#pragma unroll
    for (int r = 0; r < 8; ++r) {
      const int b = rbase + tile * 16 + r + 8 * hhalf;   // D: VGPR r -> M=r+8h
      const float c1 = ct_1[b], f1 = ft_1[b], c0 = ct[b];
      const float iu = c1 * wU0 + f1 * wU1 + c0 * wU2;
      const float ir = c1 * wR0 + f1 * wR1 + c0 * wR2;
      const float ie = c1 * wE0 + f1 * wE1 + c0 * wE2;
      const float u  = 1.f / (1.f + __expf(-(aU[r] + iu + vbu)));
      const float rt = 1.f / (1.f + __expf(-(aR[r] + ir + vbr)));
      const float e  = tanhf(rt * aE[r] + ie + vbe);
      const float hp = ht1f[(size_t)b * H + jj];
      const float hv = u * hp + (1.f - u) * e;
      ht_out[(size_t)b * H + jj] = hv;
      htb[(size_t)b * H + jj] = f2bf(hv);
    }
  }
}

// ---------------- Stage 2: fused MLP head (one branch) ----------------
// grid: BB/16, block: 128 (4 waves), one 16-row tile per block.
// y tile async-staged in LDS; hidden = relu(y@W1^T+b1) kept in LDS bf16;
// out = hidden@W2^T + b2.
#define S2_STRIDE 456   // 448 + 8 pad
__global__ __launch_bounds__(128)
void gru_stage2(const __bf16* __restrict__ yb,   // BB x H bf16, pre-offset branch*HH
                const __bf16* __restrict__ W1b,  // HH x HH bf16
                const float*  __restrict__ b1,
                const __bf16* __restrict__ W2b,  // NC x HH bf16
                const float*  __restrict__ b2,
                float* __restrict__ outp)        // BB x NC
{
  __shared__ __bf16 ytile[16 * S2_STRIDE];
  __shared__ __bf16 hid[16 * S2_STRIDE];
  const int lane  = threadIdx.x & 31;
  const int wave  = threadIdx.x >> 5;
  const int hhalf = lane >> 4;
  const int l16   = lane & 15;
  const int rbase = blockIdx.x * 16;

  // Async stage y tile: 16 rows x 56 chunks = 896 chunks, 7 per thread.
  {
    const int t = threadIdx.x;
#pragma unroll
    for (int i = 0; i < 7; ++i) {
      const int c    = t + i * 128;
      const int row  = c / 56;
      const int col8 = (c % 56) * 8;
      const unsigned loff =
          (unsigned)(uintptr_t)&ytile[row * S2_STRIDE + col8];
      async_copy_b128(loff, yb + (size_t)(rbase + row) * H + col8);
    }
  }
  wait_async0();
  __syncthreads();

  const __bf16* ap1 = ytile + (size_t)l16 * S2_STRIDE;

  // Phase 1: 28 hidden col-tiles, 7 per wave.
  for (int t = 0; t < 7; ++t) {
    const int nh = (wave * 7 + t) * 16;
    v8f acc = {};
    const __bf16* bp = W1b + (size_t)(nh + l16) * HH;
    for (int k = 0; k < HH; k += 32) {
      Frag a, bb;
      a.h[0] = *(const v8bf*)(ap1 + k + 8 * hhalf);
      a.h[1] = *(const v8bf*)(ap1 + k + 16 + 8 * hhalf);
      bb.h[0] = *(const v8bf*)(bp + k + 8 * hhalf);
      bb.h[1] = *(const v8bf*)(bp + k + 16 + 8 * hhalf);
      acc = wmma_bf16(a, bb, acc);
    }
    const float bias = b1[nh + l16];
#pragma unroll
    for (int r = 0; r < 8; ++r) {
      float v = acc[r] + bias;
      v = v > 0.f ? v : 0.f;
      hid[(r + 8 * hhalf) * S2_STRIDE + nh + l16] = f2bf(v);
    }
  }
  __syncthreads();

  // Phase 2: 16 out col-tiles, 4 per wave; A fragments from LDS.
  const __bf16* ap2 = hid + (size_t)l16 * S2_STRIDE;
  for (int t = 0; t < 4; ++t) {
    const int no = (wave * 4 + t) * 16;
    v8f acc = {};
    const __bf16* bp = W2b + (size_t)(no + l16) * HH;
    for (int k = 0; k < HH; k += 32) {
      Frag a, bb;
      a.h[0] = *(const v8bf*)(ap2 + k + 8 * hhalf);
      a.h[1] = *(const v8bf*)(ap2 + k + 16 + 8 * hhalf);
      bb.h[0] = *(const v8bf*)(bp + k + 8 * hhalf);
      bb.h[1] = *(const v8bf*)(bp + k + 16 + 8 * hhalf);
      acc = wmma_bf16(a, bb, acc);
    }
    const float bias = b2[no + l16];
#pragma unroll
    for (int r = 0; r < 8; ++r)
      outp[(size_t)(rbase + r + 8 * hhalf) * NC + no + l16] = acc[r] + bias;
  }
}

extern "C" void kernel_launch(void* const* d_in, const int* in_sizes, int n_in,
                              void* d_out, int out_size, void* d_ws, size_t ws_size,
                              hipStream_t stream) {
  const float* ht_1 = (const float*)d_in[0];
  const float* ct_1 = (const float*)d_in[1];
  const float* ft_1 = (const float*)d_in[2];
  const float* ct   = (const float*)d_in[3];
  const float* R_w  = (const float*)d_in[4];
  const float* Ic_w = (const float*)d_in[5];
  const float* If_w = (const float*)d_in[6];
  const float* O1_w = (const float*)d_in[7];
  const float* O1_b = (const float*)d_in[8];
  const float* O2_w = (const float*)d_in[9];
  const float* O2_b = (const float*)d_in[10];
  const float* O3_w = (const float*)d_in[11];
  const float* O3_b = (const float*)d_in[12];
  const float* O4_w = (const float*)d_in[13];
  const float* O4_b = (const float*)d_in[14];
  const float* bu   = (const float*)d_in[15];
  const float* br   = (const float*)d_in[16];
  const float* be   = (const float*)d_in[17];

  float* out = (float*)d_out;
  char* ws = (char*)d_ws;
  size_t off = 0;
  auto take = [&](size_t nbytes) -> char* {
    char* p = ws + off;
    off += (nbytes + 255) & ~(size_t)255;
    return p;
  };
  __bf16* ht1b = (__bf16*)take((size_t)BB * H * 2);
  __bf16* Rwb  = (__bf16*)take((size_t)3 * H * H * 2);
  __bf16* O1b  = (__bf16*)take((size_t)HH * HH * 2);
  __bf16* O2b  = (__bf16*)take((size_t)HH * HH * 2);
  __bf16* O3b  = (__bf16*)take((size_t)NC * HH * 2);
  __bf16* O4b  = (__bf16*)take((size_t)NC * HH * 2);
  __bf16* htb  = (__bf16*)take((size_t)BB * H * 2);

  auto cvt = [&](const float* s, __bf16* d, int n) {
    cvt_f32_bf16<<<(n + 255) / 256, 256, 0, stream>>>(s, d, n);
  };
  cvt(ht_1, ht1b, BB * H);
  cvt(R_w,  Rwb,  3 * H * H);
  cvt(O1_w, O1b,  HH * HH);
  cvt(O2_w, O2b,  HH * HH);
  cvt(O3_w, O3b,  NC * HH);
  cvt(O4_w, O4b,  NC * HH);

  // Stage 1: ht + bf16 copy
  dim3 g1(BB / 32, H / 64);
  gru_stage1<<<g1, 128, 0, stream>>>(ht1b, Rwb, ht_1, ct_1, ft_1, ct,
                                     Ic_w, If_w, bu, br, be, out, htb);

  // Stage 2: two branches
  float* Pct = out + (size_t)BB * H;
  float* Pft = Pct + (size_t)BB * NC;
  gru_stage2<<<BB / 16, 128, 0, stream>>>(htb,      O1b, O1_b, O3b, O3_b, Pct);
  gru_stage2<<<BB / 16, 128, 0, stream>>>(htb + HH, O2b, O2_b, O4b, O4_b, Pft);
}
